// FanChannelProcessing_8778913153137
// MI455X (gfx1250) — matrix operand order repaired and checked by reading, not compiled, not run on hardware
//
#include <hip/hip_runtime.h>
#include <hip/hip_bf16.h>
#include <float.h>
#include <stdint.h>

#define B_  32
#define S_  1024
#define C_  512
#define NH_ 8
#define HD_ 64
#define HF_ 2048

typedef __bf16 bf16;
typedef __attribute__((ext_vector_type(16))) __bf16 v16bf;
typedef __attribute__((ext_vector_type(8)))  float  v8f;

// ---------------------------------------------------------------------------
// elementwise f32 -> bf16
__global__ void __launch_bounds__(256)
k_f2b(const float* __restrict__ in, bf16* __restrict__ out) {
    size_t i = (size_t)blockIdx.x * 256 + threadIdx.x;
    out[i] = (bf16)in[i];
}

// W (K,N) f32 row-major  ->  Wt (N,K) bf16 row-major
__global__ void __launch_bounds__(256)
k_transpose_f2b(const float* __restrict__ W, bf16* __restrict__ Wt, int K, int N) {
    size_t i = (size_t)blockIdx.x * 256 + threadIdx.x;
    if (i >= (size_t)K * N) return;
    int n = (int)(i % N);
    int k = (int)(i / N);
    Wt[(size_t)n * K + k] = (bf16)W[i];
}

// ---------------------------------------------------------------------------
// C(M,N) f32 = A(M,K) bf16 row-major x Bt(N,K) bf16 row-major + bias.
// block = 256 threads = 8 waves; wave w -> rows [blockIdx.x*128 + 16w, +16),
// cols [blockIdx.y*64, +64) as 4 WMMA tiles.
// B tile (64 cols x 32 K) staged into LDS via async global->LDS copies
// (ASYNCcnt-tracked, ping-pong buffered). Main loop is 2x unrolled over K so
// the two halves alternate buffers/A-fragments with no register copies
// (avoids WMMA->VALU hazard NOPs). All 4 B fragments are pulled from LDS
// before the WMMA group so the 4 matrix ops issue back-to-back.
__global__ void __launch_bounds__(256)
gemm_bf16_wmma(const bf16* __restrict__ A, const bf16* __restrict__ Bt,
               const float* __restrict__ bias, float* __restrict__ C,
               int M, int N, int K) {   // K must be a multiple of 64
    const int wave = threadIdx.x >> 5;
    const int lane = threadIdx.x & 31;
    const bool hi  = lane >= 16;
    const int l15  = lane & 15;
    const int m0   = blockIdx.x * 128 + wave * 16;
    const int n0   = blockIdx.y * 64;

    // 80 bytes per column (32 K elems + 8 pad) -> conflict-free ds_load_b128
    __shared__ __align__(16) bf16 smB[2][64][40];

    v8f acc[4];
    #pragma unroll
    for (int t = 0; t < 4; ++t)
        #pragma unroll
        for (int r = 0; r < 8; ++r) acc[t][r] = 0.0f;

    // A fragment: lane<16 -> row m0+l15, K 0..7 & 16..23 ; lane>=16 -> K 8..15 & 24..31
    const bf16* arow = A + (size_t)(m0 + l15) * K + (hi ? 8 : 0);

    // per-thread staging slice: col = tid>>2, 16-byte segment = tid&3
    const int scol = threadIdx.x >> 2;
    const int sseg = threadIdx.x & 3;
    const bf16* gstage = Bt + (size_t)(n0 + scol) * K + sseg * 8;
    const unsigned ldsoff0 = (unsigned)(uintptr_t)&smB[0][scol][sseg * 8];
    const unsigned ldsoff1 = (unsigned)(uintptr_t)&smB[1][scol][sseg * 8];

#define STAGE_B(off, kk)                                                      \
    asm volatile("global_load_async_to_lds_b128 %0, %1, off"                  \
                 :: "v"(off), "v"(gstage + (kk)) : "memory")
#define LOAD_A(dst, kk)                                                       \
    _Pragma("unroll")                                                         \
    for (int i_ = 0; i_ < 8; ++i_) {                                          \
        (dst)[i_]     = arow[(kk) + i_];                                      \
        (dst)[i_ + 8] = arow[(kk) + 16 + i_];                                 \
    }
#define COMPUTE(bufi, afrag)                                                  \
    {                                                                         \
        v16bf bb[4];                                                          \
        _Pragma("unroll")                                                     \
        for (int t_ = 0; t_ < 4; ++t_) {                                      \
            const bf16* pb_ = &smB[bufi][t_ * 16 + l15][hi ? 8 : 0];          \
            _Pragma("unroll")                                                 \
            for (int j_ = 0; j_ < 16; ++j_) bb[t_][j_] = pb_[j_ < 8 ? j_ : j_ + 8]; \
        }                                                                     \
        _Pragma("unroll")                                                     \
        for (int t_ = 0; t_ < 4; ++t_)                                        \
            acc[t_] = __builtin_amdgcn_wmma_f32_16x16x32_bf16(                \
                false, afrag, false, bb[t_], (short)0, acc[t_], false, false);\
    }

    // prologue: stage k=0 into buffer 0, prefetch first A fragment
    STAGE_B(ldsoff0, 0);
    v16bf a0, a1;
    LOAD_A(a0, 0);

    for (int k0 = 0; k0 < K; k0 += 64) {
        // ---- half A: consume buf0 / a0, stage buf1 (k0+32 always < K) ----
        asm volatile("s_wait_asynccnt 0x0" ::: "memory");  // my buf0 writes done
        __syncthreads();                                   // everyone's done; buf1 free
        STAGE_B(ldsoff1, k0 + 32);
        LOAD_A(a1, k0 + 32);
        COMPUTE(0, a0);

        // ---- half B: consume buf1 / a1, stage buf0 for next iteration ----
        asm volatile("s_wait_asynccnt 0x0" ::: "memory");
        __syncthreads();
        if (k0 + 64 < K) {
            STAGE_B(ldsoff0, k0 + 64);
            LOAD_A(a0, k0 + 64);
        }
        COMPUTE(1, a1);
    }
#undef STAGE_B
#undef LOAD_A
#undef COMPUTE

    const int crow = m0 + (hi ? 8 : 0);
    #pragma unroll
    for (int t = 0; t < 4; ++t) {
        const int col = n0 + t * 16 + l15;
        const float bv = bias[col];
        #pragma unroll
        for (int r = 0; r < 8; ++r)
            C[(size_t)(crow + r) * N + col] = acc[t][r] + bv;
    }
}

// ---------------------------------------------------------------------------
// per-(b,c) column softmax stats over S: max and sum(exp(x - max))
__global__ void __launch_bounds__(256)
k_colstats(const float* __restrict__ X, float* __restrict__ cmax, float* __restrict__ csum) {
    const int b = blockIdx.x / C_;
    const int c = blockIdx.x % C_;
    const float* p = X + (size_t)b * S_ * C_ + c;
    const int tid = threadIdx.x;
    __shared__ float red[256];

    float m = -FLT_MAX;
    for (int s = tid; s < S_; s += 256) m = fmaxf(m, p[(size_t)s * C_]);
    red[tid] = m; __syncthreads();
    for (int o = 128; o > 0; o >>= 1) {
        if (tid < o) red[tid] = fmaxf(red[tid], red[tid + o]);
        __syncthreads();
    }
    const float mx = red[0]; __syncthreads();

    float sum = 0.0f;
    for (int s = tid; s < S_; s += 256) sum += __expf(p[(size_t)s * C_] - mx);
    red[tid] = sum; __syncthreads();
    for (int o = 128; o > 0; o >>= 1) {
        if (tid < o) red[tid] += red[tid + o];
        __syncthreads();
    }
    if (tid == 0) { cmax[blockIdx.x] = mx; csum[blockIdx.x] = red[0]; }
}

// kp[b,nh,s] = mean_hd softmax_S(k)[b,s,nh*HD+hd]
__global__ void __launch_bounds__(256)
k_kp(const float* __restrict__ X, const float* __restrict__ kmax,
     const float* __restrict__ ksum, float* __restrict__ kp) {
    const int b = blockIdx.x / S_;
    const int s = blockIdx.x % S_;
    const float* row = X + (size_t)blockIdx.x * C_;
    __shared__ float sm[C_];
    for (int c = threadIdx.x; c < C_; c += 256)
        sm[c] = __expf(row[c] - kmax[b * C_ + c]) / ksum[b * C_ + c];
    __syncthreads();
    if (threadIdx.x < NH_) {
        float acc = 0.0f;
        #pragma unroll 8
        for (int h = 0; h < HD_; ++h) acc += sm[threadIdx.x * HD_ + h];
        kp[((size_t)b * NH_ + threadIdx.x) * S_ + s] = acc * (1.0f / HD_);
    }
}

// attn[b,nh,hd] = sigmoid( sum_s softmax_S(q)[b,s,c] * kp[b,nh,s] ) * temp[nh]
__global__ void __launch_bounds__(256)
k_attn(const float* __restrict__ Q, const float* __restrict__ qmax,
       const float* __restrict__ qsum, const float* __restrict__ kp,
       const float* __restrict__ temp, float* __restrict__ attn) {
    const int b  = blockIdx.x / C_;
    const int c  = blockIdx.x % C_;
    const int nh = c / HD_;
    const float* qp  = Q  + (size_t)b * S_ * C_ + c;
    const float* kpp = kp + ((size_t)b * NH_ + nh) * S_;
    const float mx = qmax[blockIdx.x];
    const float dn = qsum[blockIdx.x];
    const int tid = threadIdx.x;
    __shared__ float red[256];

    float acc = 0.0f;
    for (int s = tid; s < S_; s += 256)
        acc += (__expf(qp[(size_t)s * C_] - mx) / dn) * kpp[s];
    red[tid] = acc; __syncthreads();
    for (int o = 128; o > 0; o >>= 1) {
        if (tid < o) red[tid] += red[tid + o];
        __syncthreads();
    }
    if (tid == 0) {
        float sig = 1.0f / (1.0f + __expf(-red[0]));
        attn[blockIdx.x] = sig * temp[nh];
    }
}

// ---------------------------------------------------------------------------
// depthwise 3x3 conv1 + exact GELU + BatchNorm (eval).  h/t layout: (b, s, hf), hf fastest.
__global__ void __launch_bounds__(256)
k_conv1_gelu_bn(const float* __restrict__ h, const float* __restrict__ w1,
                const float* __restrict__ b1, const float* __restrict__ bng,
                const float* __restrict__ bnb, const float* __restrict__ bnm,
                const float* __restrict__ bnv, float* __restrict__ t) {
    size_t idx = (size_t)blockIdx.x * 256 + threadIdx.x;   // B*S*HF exact
    const int hf = (int)(idx % HF_);
    const size_t bs = idx / HF_;
    const int s = (int)(bs % S_);
    const int b = (int)(bs / S_);
    const int y = s >> 5, x = s & 31;
    const float* wp = w1 + hf * 9;

    float acc = b1[hf];
    #pragma unroll
    for (int dy = -1; dy <= 1; ++dy) {
        const int yy = y + dy;
        if (yy < 0 || yy > 31) continue;
        #pragma unroll
        for (int dx = -1; dx <= 1; ++dx) {
            const int xx = x + dx;
            if (xx < 0 || xx > 31) continue;
            acc += wp[(dy + 1) * 3 + (dx + 1)] *
                   h[((size_t)b * S_ + ((yy << 5) + xx)) * HF_ + hf];
        }
    }
    const float g  = 0.5f * acc * (1.0f + erff(acc * 0.70710678118654752f));
    const float bn = (g - bnm[hf]) * rsqrtf(bnv[hf] + 1e-5f) * bng[hf] + bnb[hf];
    t[idx] = bn;
}

// depthwise 3x3 conv2 + mlp_weight * d + h residual -> bf16 for fc2 GEMM
__global__ void __launch_bounds__(256)
k_conv2_res(const float* __restrict__ t, const float* __restrict__ w2,
            const float* __restrict__ b2, const float* __restrict__ mlpw,
            const float* __restrict__ h, bf16* __restrict__ h2) {
    size_t idx = (size_t)blockIdx.x * 256 + threadIdx.x;
    const int hf = (int)(idx % HF_);
    const size_t bs = idx / HF_;
    const int s = (int)(bs % S_);
    const int b = (int)(bs / S_);
    const int y = s >> 5, x = s & 31;
    const float* wp = w2 + hf * 9;

    float acc = b2[hf];
    #pragma unroll
    for (int dy = -1; dy <= 1; ++dy) {
        const int yy = y + dy;
        if (yy < 0 || yy > 31) continue;
        #pragma unroll
        for (int dx = -1; dx <= 1; ++dx) {
            const int xx = x + dx;
            if (xx < 0 || xx > 31) continue;
            acc += wp[(dy + 1) * 3 + (dx + 1)] *
                   t[((size_t)b * S_ + ((yy << 5) + xx)) * HF_ + hf];
        }
    }
    h2[idx] = (bf16)(mlpw[hf] * acc + h[idx]);
}

// ---------------------------------------------------------------------------
// LayerNorm over C + gate by attn, dual-layout store
__global__ void __launch_bounds__(256)
k_ln_gate(const float* __restrict__ V, const float* __restrict__ attn,
          const float* __restrict__ lng, const float* __restrict__ lnb,
          float* __restrict__ out1, float* __restrict__ out2) {
    const int bs = blockIdx.x;
    const int b = bs / S_;
    const int s = bs % S_;
    const float* row = V + (size_t)bs * C_;
    const int tid = threadIdx.x;
    __shared__ float r0[256], r1[256];

    const float x0 = row[tid];
    const float x1 = row[tid + 256];
    r0[tid] = x0 + x1;
    r1[tid] = x0 * x0 + x1 * x1;
    __syncthreads();
    for (int o = 128; o > 0; o >>= 1) {
        if (tid < o) { r0[tid] += r0[tid + o]; r1[tid] += r1[tid + o]; }
        __syncthreads();
    }
    const float mu  = r0[0] * (1.0f / C_);
    const float var = r1[0] * (1.0f / C_) - mu * mu;
    const float rs  = rsqrtf(var + 1e-6f);

    #pragma unroll
    for (int half = 0; half < 2; ++half) {
        const int c = tid + half * 256;
        const float xv = (half == 0) ? x0 : x1;
        const float xn = (xv - mu) * rs * lng[c] + lnb[c];
        const float g  = attn[b * C_ + c] * xn;
        out1[(size_t)bs * C_ + c] = g;
        const int nh = c / HD_, hd = c % HD_;
        out2[(((size_t)b * NH_ + nh) * S_ + s) * HD_ + hd] = g;
    }
}

// ---------------------------------------------------------------------------
extern "C" void kernel_launch(void* const* d_in, const int* in_sizes, int n_in,
                              void* d_out, int out_size, void* d_ws, size_t ws_size,
                              hipStream_t stream) {
    const float* x      = (const float*)d_in[0];
    const float* q_w    = (const float*)d_in[3];
    const float* q_b    = (const float*)d_in[4];
    const float* temp   = (const float*)d_in[5];
    const float* fc1_w  = (const float*)d_in[6];
    const float* fc1_b  = (const float*)d_in[7];
    const float* fc2_w  = (const float*)d_in[8];
    const float* fc2_b  = (const float*)d_in[9];
    const float* mlpw   = (const float*)d_in[10];
    const float* c1_w   = (const float*)d_in[11];
    const float* c1_b   = (const float*)d_in[12];
    const float* c2_w   = (const float*)d_in[13];
    const float* c2_b   = (const float*)d_in[14];
    const float* bng    = (const float*)d_in[15];
    const float* bnb    = (const float*)d_in[16];
    const float* bnm    = (const float*)d_in[17];
    const float* bnv    = (const float*)d_in[18];
    const float* lng    = (const float*)d_in[19];
    const float* lnb    = (const float*)d_in[20];

    float* out1 = (float*)d_out;
    float* out2 = out1 + (size_t)B_ * S_ * C_;

    // workspace carve-up
    char* ws = (char*)d_ws;
    size_t off = 0;
    auto alloc = [&](size_t bytes) { void* p = ws + off; off = (off + bytes + 255) & ~(size_t)255; return p; };
    bf16*  xb   = (bf16*) alloc((size_t)B_ * S_ * C_  * sizeof(bf16));
    bf16*  wqT  = (bf16*) alloc((size_t)C_ * C_       * sizeof(bf16));
    bf16*  w1T  = (bf16*) alloc((size_t)HF_ * C_      * sizeof(bf16));
    bf16*  w2T  = (bf16*) alloc((size_t)C_ * HF_      * sizeof(bf16));
    float* q    = (float*)alloc((size_t)B_ * S_ * C_  * sizeof(float));
    float* h    = (float*)alloc((size_t)B_ * S_ * HF_ * sizeof(float));
    float* t    = (float*)alloc((size_t)B_ * S_ * HF_ * sizeof(float));
    bf16*  h2b  = (bf16*) alloc((size_t)B_ * S_ * HF_ * sizeof(bf16));
    float* v    = (float*)alloc((size_t)B_ * S_ * C_  * sizeof(float));
    float* kmax = (float*)alloc((size_t)B_ * C_ * sizeof(float));
    float* ksum = (float*)alloc((size_t)B_ * C_ * sizeof(float));
    float* qmax = (float*)alloc((size_t)B_ * C_ * sizeof(float));
    float* qsum = (float*)alloc((size_t)B_ * C_ * sizeof(float));
    float* kp   = (float*)alloc((size_t)B_ * NH_ * S_ * sizeof(float));
    float* attn = (float*)alloc((size_t)B_ * C_ * sizeof(float));
    (void)in_sizes; (void)n_in; (void)out_size; (void)ws_size;

    const int M = B_ * S_;                 // 32768

    // 1) precision conversions / weight transposes
    k_f2b<<<(B_ * S_ * C_) / 256, 256, 0, stream>>>(x, xb);
    k_transpose_f2b<<<(C_ * C_ + 255) / 256, 256, 0, stream>>>(q_w,   wqT, C_,  C_);
    k_transpose_f2b<<<(C_ * HF_ + 255) / 256, 256, 0, stream>>>(fc1_w, w1T, C_,  HF_);
    k_transpose_f2b<<<(HF_ * C_ + 255) / 256, 256, 0, stream>>>(fc2_w, w2T, HF_, C_);

    // 2) q = x @ q_w + q_b
    gemm_bf16_wmma<<<dim3(M / 128, C_ / 64), 256, 0, stream>>>(xb, wqT, q_b, q, M, C_, C_);

    // 3) k path: softmax stats on x, then kp
    k_colstats<<<B_ * C_, 256, 0, stream>>>(x, kmax, ksum);
    k_kp<<<B_ * S_, 256, 0, stream>>>(x, kmax, ksum, kp);

    // 4) attn: softmax stats on q, dot with kp, sigmoid * temperature
    k_colstats<<<B_ * C_, 256, 0, stream>>>(q, qmax, qsum);
    k_attn<<<B_ * C_, 256, 0, stream>>>(q, qmax, qsum, kp, temp, attn);

    // 5) h = x @ fc1_w + fc1_b
    gemm_bf16_wmma<<<dim3(M / 128, HF_ / 64), 256, 0, stream>>>(xb, w1T, fc1_b, h, M, HF_, C_);

    // 6) depthwise conv1 + gelu + bn ; conv2 + residual (bf16 out)
    k_conv1_gelu_bn<<<(size_t)(B_ * S_) * HF_ / 256, 256, 0, stream>>>(
        h, c1_w, c1_b, bng, bnb, bnm, bnv, t);
    k_conv2_res<<<(size_t)(B_ * S_) * HF_ / 256, 256, 0, stream>>>(
        t, c2_w, c2_b, mlpw, h, h2b);

    // 7) v = h2 @ fc2_w + fc2_b
    gemm_bf16_wmma<<<dim3(M / 128, C_ / 64), 256, 0, stream>>>(h2b, w2T, fc2_b, v, M, C_, HF_);

    // 8) LayerNorm + gate, dual-layout outputs
    k_ln_gate<<<B_ * S_, 256, 0, stream>>>(v, attn, lng, lnb, out1, out2);
}
